// Net_58523224375973
// MI455X (gfx1250) — compile-verified, matrix operand order checked by reading, not run on hardware
//
#include <hip/hip_runtime.h>
#include <hip/hip_bf16.h>

// Problem constants (from reference)
#define N_HID   512
#define BATCH   256
#define TSTEPS  512
#define KCHUNKS 16                 // K=512 in chunks of 32
#define ALPHA_C 0.2f
#define IN_SCALE 0.6324555320336759f   // sqrt(2/alpha * sigma_in^2) = sqrt(0.4)

typedef __attribute__((ext_vector_type(16))) __bf16 v16bf;
typedef __attribute__((ext_vector_type(8)))  float  v8f;
typedef int v4i __attribute__((vector_size(4 * sizeof(int))));
typedef __attribute__((address_space(1))) v4i gv4i;   // global 16B vector
typedef __attribute__((address_space(3))) v4i lv4i;   // LDS 16B vector

// 16B global -> LDS copy through the async engine (ASYNCcnt, no VGPR data).
__device__ __forceinline__ void async_b128(const char* g, char* l) {
  __builtin_amdgcn_global_load_async_to_lds_b128((gv4i*)g, (lv4i*)l, 0, 0);
}

// s_wait_asynccnt needs a source-level constant; switch folds after unroll.
__device__ __forceinline__ void wait_asynccnt(int n) {
  switch (n) {
    case 0:  __builtin_amdgcn_s_wait_asynccnt(0);  break;
    case 8:  __builtin_amdgcn_s_wait_asynccnt(8);  break;
    case 16: __builtin_amdgcn_s_wait_asynccnt(16); break;
    default: __builtin_amdgcn_s_wait_asynccnt(24); break;
  }
}

// ---------------------------------------------------------------------------
// Fragment layout (CDNA5 WMMA bf16, wave32):
//  A (16x32, MxK): lanes 0-15 hold rows M=0..15 with K={0..7,16..23} packed
//  2/VGPR; lanes 16-31 hold same rows with K={8..15,24..31}.
//  B (32x16, KxN): mirrored — lane = N + 16*(K-group), 16 bf16 (32B) per lane.
//  For byte-pair index j (0..15) within a lane's 32B fragment and lane-half
//  bit hi: k_local = (j&7) + 16*(j>>3) + 8*hi.
// ---------------------------------------------------------------------------

// Pre-swizzle W_rec (f32, row-major [n][k]) into bf16 B-fragments so each
// lane's fragment is 32 contiguous bytes. Also zero out[:,0,:].
__global__ void prep_kernel(const float* __restrict__ Wrec,
                            __bf16* __restrict__ wtiles,
                            float* __restrict__ out) {
  int idx = blockIdx.x * blockDim.x + threadIdx.x;
  if (idx < N_HID * N_HID) {
    int j    = idx & 15;          // bf16 index within lane fragment
    int lane = (idx >> 4) & 31;   // lane within wave
    int blk  = idx >> 9;          // fragment block = nt*KCHUNKS + kc
    int kc   = blk & (KCHUNKS - 1);
    int nt   = blk >> 4;
    int hi   = lane >> 4;
    int n    = nt * 16 + (lane & 15);
    int klocal = (j & 7) + ((j >> 3) << 4) + (hi << 3);
    int k    = kc * 32 + klocal;
    // B[k][n] = W_rec[n][k]  (matmul is relu(s) @ W_rec^T)
    wtiles[idx] = (__bf16)Wrec[n * N_HID + k];
  }
  if (idx < BATCH * N_HID) {      // states[:,0,:] = 0
    int b = idx >> 9;
    int n = idx & (N_HID - 1);
    out[(size_t)b * TSTEPS * N_HID + n] = 0.0f;
  }
}

// One workgroup = 16 batch rows, full N=512 state held in registers across all
// timesteps. 8 waves x 4 N-tiles each. Per step per wave: 64 wmma.
// W (512 KB bf16, L2-resident) is streamed into 4 rotating LDS stages by the
// async engine, 3 chunks ahead of consumption.
__global__ __launch_bounds__(256)
void rnn_scan_kernel(const float* __restrict__ u,
                     const float* __restrict__ inoise,
                     const float* __restrict__ rnoise,
                     const float* __restrict__ Win,
                     const float* __restrict__ brec,
                     const __bf16* __restrict__ wtiles,
                     float* __restrict__ out) {
  // relu(s) A-fragments, double buffered: 2*16*32*32B = 32 KB
  __shared__ __align__(32) __bf16 afrag_lds[2][KCHUNKS][32][16];
  // B-fragment stream: 4 stages x 8 waves x 4 frags x 1 KB = 128 KB
  __shared__ __align__(32) char bstage_lds[4][8][4][1024];

  const int lane   = threadIdx.x & 31;
  const int wave   = threadIdx.x >> 5;   // 0..7 (wave32)
  const int hi     = lane >> 4;
  const int lo     = lane & 15;
  const int b_base = blockIdx.x * 16;

  const v8f zerov = {0.f, 0.f, 0.f, 0.f, 0.f, 0.f, 0.f, 0.f};

  // Loop-invariant per-lane coefficients (12 scalars in VGPRs for all steps)
  float win0[4], win1[4], bias[4];
  #pragma unroll
  for (int i = 0; i < 4; ++i) {
    const int n = ((wave << 2) + i) * 16 + lo;
    win0[i] = Win[n * 2 + 0];
    win1[i] = Win[n * 2 + 1];
    bias[i] = brec[n];
  }

  // State registers: sreg[i][v] = s[m = v + 8*hi][n = (4*wave+i)*16 + lo]
  v8f sreg[4];
  #pragma unroll
  for (int i = 0; i < 4; ++i) sreg[i] = zerov;

  // Per-lane W addressing: wave*64KB + lane*32B
  const unsigned wo = (unsigned)(wave << 16) + (unsigned)(lane << 5);
  const char* gbase = (const char*)wtiles + wo;
  char* lbase = (char*)&bstage_lds[0][wave][0][0] + (lane << 5);

  // Issue one stream chunk: wave's 4 B-fragments (4 KB) for k-chunk kc into
  // LDS stage kc&3 — 8 async b128 ops, wave-private (no barrier needed).
  auto issue_chunk = [&](int kc) {
    const int st = kc & 3;
    #pragma unroll
    for (int i = 0; i < 4; ++i) {
      const char* g = gbase + (i * 16384 + kc * 1024);
      char*       l = lbase + (st * 32768 + i * 1024);
      async_b128(g, l);
      async_b128(g + 16, l + 16);
    }
  };

  for (int t = 0; t < TSTEPS - 1; ++t) {
    const int buf = t & 1;

    // Prime the stream 3 chunks deep right at step start (prev step fully
    // drained ASYNCcnt at its kc=15, and all stage reads were waited-on by
    // the wmma that consumed them, so stage reuse is WAR-safe).
    issue_chunk(0); issue_chunk(1); issue_chunk(2);

    // ---- Phase 0: kick off HBM-bound operands; latency hides under WMMA.
    float uu0[8], uu1[8], rn[8][4];
    #pragma unroll
    for (int v = 0; v < 8; ++v) {
      const int m = v + (hi << 3);
      const int b = b_base + m;
      const size_t ub = ((size_t)b * TSTEPS + t) * 2;
      uu0[v] = u[ub + 0] + IN_SCALE * inoise[ub + 0];
      uu1[v] = u[ub + 1] + IN_SCALE * inoise[ub + 1];
      const size_t rbase = ((size_t)b * TSTEPS + t) * N_HID;
      #pragma unroll
      for (int i = 0; i < 4; ++i) {
        const int n = ((wave << 2) + i) * 16 + lo;
        rn[v][i] = rnoise[rbase + n];
      }
      __builtin_prefetch(&rnoise[rbase + N_HID + ((wave << 6) + lo)], 0, 0);
    }

    // ---- Phase 1: scatter relu(s) as bf16 into A-fragment layout ----
    #pragma unroll
    for (int v = 0; v < 8; ++v) {
      const int m = v + (hi << 3);
      #pragma unroll
      for (int i = 0; i < 4; ++i) {
        const int n  = ((wave << 2) + i) * 16 + lo;   // becomes K next matmul
        float r = fmaxf(sreg[i][v], 0.0f);
        const int kc     = n >> 5;
        const int klocal = n & 31;
        const int lane2  = m + (((klocal >> 3) & 1) << 4);
        const int j      = (klocal & 7) + ((klocal >> 4) << 3);
        afrag_lds[buf][kc][lane2][j] = (__bf16)r;
      }
    }
    __syncthreads();   // single barrier/step (double-buffered A LDS)

    // ---- Phase 2: relu(s)[16x512] @ W^T[512x512] via WMMA ----
    v8f acc[4];
    #pragma unroll
    for (int i = 0; i < 4; ++i) acc[i] = zerov;

    #pragma unroll
    for (int kc = 0; kc < KCHUNKS; ++kc) {
      if (kc + 3 < KCHUNKS) issue_chunk(kc + 3);
      // Outstanding after this point: chunks kc..kc+3 (<=32 ops); wait until
      // kc's 8 ops retired (async loads retire in order).
      wait_asynccnt((kc <= 12) ? 24 : 8 * (15 - kc));
      asm volatile("" ::: "memory");     // keep ds reads below the wait

      v16bf a = *(const v16bf*)(&afrag_lds[buf][kc][lane][0]);  // 2x ds_load_b128
      v16bf bf[4];
      #pragma unroll
      for (int i = 0; i < 4; ++i) {
        bf[i] = *(const v16bf*)(lbase + ((kc & 3) * 32768 + i * 1024));
      }
      #pragma unroll
      for (int i = 0; i < 4; ++i) {
        acc[i] = __builtin_amdgcn_wmma_f32_16x16x32_bf16(
            /*neg_a=*/false, a, /*neg_b=*/false, bf[i],
            /*c_mod=*/(short)0, acc[i], /*reuse_a=*/false, /*reuse_b=*/false);
      }
    }

    // ---- Phase 3: leaky update from pre-loaded operands, store state ----
    #pragma unroll
    for (int v = 0; v < 8; ++v) {
      const int m = v + (hi << 3);
      const int b = b_base + m;
      const size_t obase = ((size_t)b * TSTEPS + (t + 1)) * N_HID;
      #pragma unroll
      for (int i = 0; i < 4; ++i) {
        const int n = ((wave << 2) + i) * 16 + lo;
        const float drive = uu0[v] * win0[i] + uu1[v] * win1[i]
                          + rn[v][i] + bias[i];
        const float sv    = sreg[i][v];
        const float val   = (1.0f - ALPHA_C) * sv + ALPHA_C * (acc[i][v] + drive);
        sreg[i][v] = val;
        out[obase + n] = val;    // coalesced 64B per half-wave
      }
    }
  }
}

extern "C" void kernel_launch(void* const* d_in, const int* in_sizes, int n_in,
                              void* d_out, int out_size, void* d_ws, size_t ws_size,
                              hipStream_t stream) {
  const float* u      = (const float*)d_in[0];
  const float* inoise = (const float*)d_in[1];
  const float* rnoise = (const float*)d_in[2];
  const float* Win    = (const float*)d_in[3];
  const float* Wrec   = (const float*)d_in[4];
  const float* brec   = (const float*)d_in[5];
  float* out          = (float*)d_out;
  __bf16* wtiles      = (__bf16*)d_ws;   // 512*512*2 = 512 KB of scratch

  prep_kernel<<<(N_HID * N_HID + 255) / 256, 256, 0, stream>>>(Wrec, wtiles, out);
  // 16 workgroups x 256 threads: one 16-row batch slice per WGP, serial in T
  rnn_scan_kernel<<<BATCH / 16, 256, 0, stream>>>(u, inoise, rnoise, Win, brec,
                                                  wtiles, out);
}
